// DGCNN_PartSeg_841813590605
// MI455X (gfx1250) — matrix operand order repaired and checked by reading, not compiled, not run on hardware
//
// DGCNN part-seg forward for MI455X (gfx1250, wave32, WMMA).
// Heavy GEMMs (1x1 convs) -> v_wmma_f32_16x16x32_f16, f32 accumulate.
// Gathers / max-reductions fused into GEMM operands / epilogues; O(N^2)
// pair matrix, [B,128,N,k] edge tensor and [B,1024,N] conv4 output are
// never materialized. Fragment construction is branch-free (clamped
// indices + selects, no EXEC divergence); f16 GEMMs stage B panels in
// LDS via async global->LDS loads when available.

#include <hip/hip_runtime.h>

namespace {

constexpr int BB   = 16;      // batch
constexpr int NN   = 2048;    // points
constexpr int KK   = 20;      // knn neighbors
constexpr int BN   = BB * NN;          // 32768
constexpr int MTK  = BB * NN * KK;     // 655360 GEMM columns for edge convs
constexpr float EPS   = 1e-5f;
constexpr float LRELU = 0.2f;

typedef __attribute__((ext_vector_type(16))) _Float16 v16h;
typedef __attribute__((ext_vector_type(8)))  float    v8f;

#if __has_builtin(__builtin_amdgcn_global_load_async_to_lds_b32)
#define HAVE_ASYNC_LDS 1
typedef __attribute__((address_space(1))) int gas_int;
typedef __attribute__((address_space(3))) int lds_int;
#else
#define HAVE_ASYNC_LDS 0
#endif

__device__ __forceinline__ void wait_async_then_barrier() {
#if HAVE_ASYNC_LDS
#if __has_builtin(__builtin_amdgcn_s_wait_asynccnt)
  __builtin_amdgcn_s_wait_asynccnt(0);
#else
  asm volatile("s_wait_asynccnt 0x0" ::: "memory");
#endif
#endif
  __syncthreads();
}

// ---------- helpers ----------
__device__ __forceinline__ float bn_act(float y, const float* bn, int C, int c) {
  float g = bn[c], b = bn[C + c], m = bn[2 * C + c], v = bn[3 * C + c];
  y = g * (y - m) * rsqrtf(v + EPS) + b;
  return y > 0.f ? y : LRELU * y;
}
// order-preserving float->uint key (for max via integer atomicMax)
__device__ __forceinline__ unsigned fkey(float f) {
  unsigned u = __float_as_uint(f);
  return (u & 0x80000000u) ? ~u : (u | 0x80000000u);
}
__device__ __forceinline__ float kinv(unsigned k) {
  return __uint_as_float((k & 0x80000000u) ? (k ^ 0x80000000u) : ~k);
}
// k order chosen here (contiguous, not AMD's table; A and B match): A row M=lane%16, K = h (+16 on hi
// lanes). B-operand: column = lane%16, K = h (+16 on hi lanes).
__device__ __forceinline__ int a_kmap(int lane, int h) {
  return ((lane & 16) ? 16 : 0) + h;
}
__device__ __forceinline__ int b_kmap(int lane, int h) {
  return ((lane & 16) ? 16 : 0) + h;
}

// ---------- kNN: one block per (b,n) row; distances in LDS, 20 argmax rounds
__global__ void knn_kernel(const float* __restrict__ feat, long bstride, int C,
                           int* __restrict__ idx) {
  __shared__ float dist[NN];
  __shared__ float ctr[64];
  __shared__ unsigned long long red[256];
  const int b = blockIdx.x / NN, n = blockIdx.x % NN;
  const float* fb = feat + (long)b * bstride;
  for (int c = threadIdx.x; c < C; c += 256) ctr[c] = fb[c * NN + n];
  __syncthreads();
  for (int m = threadIdx.x; m < NN; m += 256) {
    float s = 0.f;
    for (int c = 0; c < C; ++c) { float d = ctr[c] - fb[c * NN + m]; s -= d * d; }
    dist[m] = s;  // == 2<x_n,x_m> - |x_n|^2 - |x_m|^2
  }
  __syncthreads();
  for (int j = 0; j < KK; ++j) {
    unsigned long long best = 0ull;
    for (int m = threadIdx.x; m < NN; m += 256) {
      unsigned long long key =
          ((unsigned long long)fkey(dist[m]) << 32) | (unsigned)(0x7fffffff - m);
      if (key > best) best = key;
    }
    red[threadIdx.x] = best;
    __syncthreads();
    for (int s = 128; s > 0; s >>= 1) {
      if (threadIdx.x < s && red[threadIdx.x + s] > red[threadIdx.x])
        red[threadIdx.x] = red[threadIdx.x + s];
      __syncthreads();
    }
    int sel = 0x7fffffff - (int)(unsigned)(red[0] & 0xffffffffu);
    if (threadIdx.x == 0) {
      idx[((long)b * NN + n) * KK + j] = sel;
      dist[sel] = -3.4e38f;
    }
    __syncthreads();
  }
}

// ---------- edge conv (conv1a / conv2a / conv3): gather fused into B operand.
// out[co][m] = f16(bn_act(sum_c W[co,c]*edge[c,m])), m = ((b*N+n)*K+kk).
// Branch-free operand build: unconditional clamped loads + value selects.
__global__ void conv_edge_gemm(const float* __restrict__ W,
                               const float* __restrict__ bn,
                               const float* __restrict__ feat, long bstride,
                               const int* __restrict__ idx,
                               _Float16* __restrict__ out,
                               int Co, int Ci, int chalf) {
  const int lane = threadIdx.x & 31;
  const int wave = blockIdx.x * (blockDim.x >> 5) + (threadIdx.x >> 5);
  const int tilesRow = Co >> 4;
  if (wave >= tilesRow * (MTK >> 4)) return;   // wave-uniform: EXEC stays full
  const int tr = wave % tilesRow, tc = wave / tilesRow;
  const int co0 = tr * 16;
  const int m = tc * 16 + (lane & 15);
  const int b = m / (NN * KK);
  const int rem = m - b * (NN * KK);
  const int n = rem / KK;
  const int nb = idx[m];
  const float* fb = feat + (long)b * bstride;
  const int arow = co0 + (lane & 15);

  v8f acc = {};
  for (int kb = 0; kb < Ci; kb += 32) {
    float wv[16], p1[16], p2[16];
#pragma unroll
    for (int h = 0; h < 16; ++h) {
      int ka = kb + a_kmap(lane, h);
      wv[h] = W[arow * Ci + (ka < Ci ? ka : 0)];
      int kx = kb + b_kmap(lane, h);
      bool lo = kx < chalf, ok = kx < Ci;
      int c = lo ? kx : (ok ? kx - chalf : 0);
      p1[h] = fb[c * NN + (lo ? nb : n)];   // neighbor (or center) term
      p2[h] = fb[c * NN + n];               // center term for the diff half
    }
    v16h a, bx;
#pragma unroll
    for (int h = 0; h < 16; ++h) {
      int ka = kb + a_kmap(lane, h);
      a[h] = (_Float16)(ka < Ci ? wv[h] : 0.f);
      int kx = kb + b_kmap(lane, h);
      bool lo = kx < chalf, ok = kx < Ci;
      bx[h] = (_Float16)(ok ? (lo ? p1[h] - p2[h] : p1[h]) : 0.f);
    }
    acc = __builtin_amdgcn_wmma_f32_16x16x32_f16(false, a, false, bx,
                                                 (short)0, acc, false, false);
  }
  const int rbase = co0 + ((lane & 16) ? 8 : 0);
#pragma unroll
  for (int r = 0; r < 8; ++r) {
    int co = rbase + r;
    out[(long)co * MTK + m] = (_Float16)bn_act(acc[r], bn, Co, co);
  }
}

// ---------- f16-activation GEMM (conv1b/2b mode0 over MTK; conv6/7 mode0,
// conv8 mode1 over BN). One block per 16-column panel; the panel (Ci x 16
// halves) is staged in LDS (async global->LDS when available), then the
// 8 waves sweep the Co/16 row tiles with B fragments read from LDS.
__global__ void conv_f16_gemm(const float* __restrict__ W,
                              const float* __restrict__ bn,
                              const _Float16* __restrict__ X,
                              _Float16* __restrict__ out,
                              float* __restrict__ out32,
                              const float* __restrict__ bias,
                              int Co, int Ci, int Mtot, int mode) {
  __shared__ _Float16 Bp[256 * 16];  // [k][col], up to 8 KB
  const int tid = threadIdx.x;
  const int m0 = blockIdx.x * 16;

  {  // stage B panel, u32 granularity, coalesced: row k = 16 halves = 8 u32
    const unsigned* Xu = (const unsigned*)(X + m0);
    unsigned* Bu = (unsigned*)Bp;
    const long rowU = (long)Mtot >> 1;
    for (int i = tid; i < Ci * 8; i += 256) {
      int k = i >> 3, cp = i & 7;
#if HAVE_ASYNC_LDS
      __builtin_amdgcn_global_load_async_to_lds_b32(
          (gas_int*)(Xu + (long)k * rowU + cp), (lds_int*)(Bu + i), 0, 0);
#else
      Bu[i] = Xu[(long)k * rowU + cp];
#endif
    }
  }
  wait_async_then_barrier();

  const int lane = tid & 31;
  const int col  = lane & 15;
  const int tilesRow = (Co + 15) >> 4;
  for (int tr = tid >> 5; tr < tilesRow; tr += 8) {  // wave-uniform loop
    const int co0 = tr * 16;
    const int arow = co0 + col;
    const int arc = arow < Co ? arow : 0;
    v8f acc = {};
    for (int kb = 0; kb < Ci; kb += 32) {
      v16h a, bx;
#pragma unroll
      for (int h = 0; h < 16; ++h) {
        int ka = kb + a_kmap(lane, h);
        float w = W[arc * Ci + ka];
        a[h] = (_Float16)(arow < Co ? w : 0.f);
        bx[h] = Bp[(kb + b_kmap(lane, h)) * 16 + col];
      }
      acc = __builtin_amdgcn_wmma_f32_16x16x32_f16(false, a, false, bx,
                                                   (short)0, acc, false, false);
    }
    const int rbase = co0 + ((lane & 16) ? 8 : 0);
    const int m = m0 + col;
    if (mode == 0) {
#pragma unroll
      for (int r = 0; r < 8; ++r) {
        int co = rbase + r;
        out[(long)co * Mtot + m] = (_Float16)bn_act(acc[r], bn, Co, co);
      }
    } else {  // conv8: + bias, f32 [B,Co,N] output, Co ragged (50)
      int b = m / NN, n = m % NN;
#pragma unroll
      for (int r = 0; r < 8; ++r) {
        int co = rbase + r;
        if (co < Co) out32[((long)b * Co + co) * NN + n] = acc[r] + bias[co];
      }
    }
  }
}

// ---------- GEMM over x123 ([B,192,N] f32): conv4+max fusion / conv5+bias
__global__ void conv_x123_gemm(const float* __restrict__ W, int Wld, int Wcoff,
                               const float* __restrict__ bn,
                               const float* __restrict__ x123,
                               int Co, int mode,
                               unsigned* __restrict__ xmaxU,
                               const float* __restrict__ bias5,
                               _Float16* __restrict__ out) {
  const int lane = threadIdx.x & 31;
  const int wave = blockIdx.x * (blockDim.x >> 5) + (threadIdx.x >> 5);
  const int tilesRow = Co >> 4;
  if (wave >= tilesRow * (BN >> 4)) return;
  const int tr = wave % tilesRow, tc = wave / tilesRow;
  const int co0 = tr * 16;
  const int m = tc * 16 + (lane & 15);
  const int b = m / NN, n = m % NN;
  const float* xb = x123 + (long)b * 192 * NN + n;
  const int arow = co0 + (lane & 15);

  v8f acc = {};
  for (int kb = 0; kb < 192; kb += 32) {
    v16h a, bx;
#pragma unroll
    for (int h = 0; h < 16; ++h) {
      int ka = kb + a_kmap(lane, h);
      a[h] = (_Float16)W[arow * Wld + Wcoff + ka];
      int kx = kb + b_kmap(lane, h);
      bx[h] = (_Float16)xb[(long)kx * NN];
    }
    acc = __builtin_amdgcn_wmma_f32_16x16x32_f16(false, a, false, bx,
                                                 (short)0, acc, false, false);
  }
  const int rbase = co0 + ((lane & 16) ? 8 : 0);
#pragma unroll
  for (int r = 0; r < 8; ++r) {
    int co = rbase + r;
    if (mode == 0) {  // conv4: bn_act then max over n (order-preserving uint)
      float y = bn_act(acc[r], bn, Co, co);
      atomicMax(&xmaxU[b * 1024 + co], fkey(y));
    } else {          // conv5: + hoisted bias (1088 n-invariant channels)
      float y = bn_act(acc[r] + bias5[b * 256 + co], bn, Co, co);
      out[(long)co * BN + m] = (_Float16)y;
    }
  }
}

// ---------- max over k (after bn_act, matching reference order)
__global__ void max_over_k(const _Float16* __restrict__ e,
                           float* __restrict__ x123, int cbase) {
  long t = (long)blockIdx.x * blockDim.x + threadIdx.x;
  if (t >= (long)64 * BN) return;
  int c = (int)(t / BN);
  int r = (int)(t % BN);
  int b = r / NN, n = r % NN;
  const _Float16* p = e + (long)c * MTK + (long)r * KK;
  float mx = (float)p[0];
  for (int i = 1; i < KK; ++i) { float v = (float)p[i]; mx = v > mx ? v : mx; }
  x123[((long)b * 192 + cbase + c) * NN + n] = mx;
}

__global__ void init_xmax(unsigned* __restrict__ xmaxU) {
  int t = blockIdx.x * blockDim.x + threadIdx.x;
  if (t < BB * 1024) xmaxU[t] = 0u;  // 0 < fkey(x) for all finite x
}

__global__ void lv_kernel(const float* __restrict__ Wl,
                          const float* __restrict__ bnl,
                          const float* __restrict__ l, float* __restrict__ lv) {
  int t = blockIdx.x * blockDim.x + threadIdx.x;
  if (t >= BB * 64) return;
  int b = t / 64, c = t % 64;
  float s = 0.f;
  for (int i = 0; i < 16; ++i) s += Wl[c * 16 + i] * l[b * 16 + i];
  lv[t] = bn_act(s, bnl, 64, c);
}

// bias5[b][o] = W5[:, :1024] @ xmax + W5[:, 1024:1088] @ lv  (n-invariant part)
__global__ void bias5_kernel(const float* __restrict__ W5,
                             const unsigned* __restrict__ xmaxU,
                             const float* __restrict__ lv,
                             float* __restrict__ bias5) {
  int t = blockIdx.x * blockDim.x + threadIdx.x;
  if (t >= BB * 256) return;
  int b = t / 256, o = t % 256;
  float s = 0.f;
  for (int c = 0; c < 1024; ++c) s += W5[o * 1280 + c] * kinv(xmaxU[b * 1024 + c]);
  for (int c = 0; c < 64; ++c)  s += W5[o * 1280 + 1024 + c] * lv[b * 64 + c];
  bias5[t] = s;
}

}  // namespace

extern "C" void kernel_launch(void* const* d_in, const int* in_sizes, int n_in,
                              void* d_out, int out_size, void* d_ws, size_t ws_size,
                              hipStream_t stream) {
  const float* x    = (const float*)d_in[0];
  const float* l    = (const float*)d_in[1];
  const float* W1a  = (const float*)d_in[2];  const float* bn1a = (const float*)d_in[3];
  const float* W1b  = (const float*)d_in[4];  const float* bn1b = (const float*)d_in[5];
  const float* W2a  = (const float*)d_in[6];  const float* bn2a = (const float*)d_in[7];
  const float* W2b  = (const float*)d_in[8];  const float* bn2b = (const float*)d_in[9];
  const float* W3   = (const float*)d_in[10]; const float* bn3  = (const float*)d_in[11];
  const float* W4   = (const float*)d_in[12]; const float* bn4  = (const float*)d_in[13];
  const float* Wl   = (const float*)d_in[14]; const float* bnl  = (const float*)d_in[15];
  const float* W5   = (const float*)d_in[16]; const float* bn5  = (const float*)d_in[17];
  const float* W6   = (const float*)d_in[18]; const float* bn6  = (const float*)d_in[19];
  const float* W7   = (const float*)d_in[20]; const float* bn7  = (const float*)d_in[21];
  const float* W8   = (const float*)d_in[22]; const float* b8   = (const float*)d_in[23];
  float* out = (float*)d_out;

  // workspace layout (bytes); h buffers are reused for the y1/y2/y3 stages
  char* ws = (char*)d_ws;
  int*      idx   = (int*)(ws + 0);                         //  2,621,440
  float*    x123  = (float*)(ws + 2621440);                 // 25,165,824 [B,192,N]
  unsigned* xmaxU = (unsigned*)(ws + 27787264);             //     65,536
  float*    lv    = (float*)(ws + 27852800);                //      4,096
  float*    bias5 = (float*)(ws + 27856896);                //     16,384
  _Float16* hA    = (_Float16*)(ws + 27873280);             // 83,886,080 [64,MTK]
  _Float16* hB    = (_Float16*)(ws + 111759360);            // 83,886,080
  _Float16* y1 = hA; _Float16* y2 = hB; _Float16* y3 = hA;  // [C, B*N] f16

  const int TPB = 256;
  const int edgeBlk = (4 * (MTK / 16)) / 8;       // 20480 (Co=64 edge convs)
  const int maxBlk  = (64 * BN + TPB - 1) / TPB;  // 8192
  const int panMTK = MTK / 16;                    // 40960 panels
  const int panBN  = BN / 16;                     // 2048 panels

  // ---- stage 1 (C=3 kNN, conv1a 6->64, conv1b 64->64, max_k -> x1)
  knn_kernel<<<BN, TPB, 0, stream>>>(x, (long)3 * NN, 3, idx);
  conv_edge_gemm<<<edgeBlk, TPB, 0, stream>>>(W1a, bn1a, x, (long)3 * NN, idx, hA, 64, 6, 3);
  conv_f16_gemm<<<panMTK, TPB, 0, stream>>>(W1b, bn1b, hA, hB, nullptr, nullptr, 64, 64, MTK, 0);
  max_over_k<<<maxBlk, TPB, 0, stream>>>(hB, x123, 0);

  // ---- stage 2 (C=64 kNN on x1, conv2a 128->64, conv2b 64->64 -> x2)
  knn_kernel<<<BN, TPB, 0, stream>>>(x123, (long)192 * NN, 64, idx);
  conv_edge_gemm<<<edgeBlk, TPB, 0, stream>>>(W2a, bn2a, x123, (long)192 * NN, idx, hA, 64, 128, 64);
  conv_f16_gemm<<<panMTK, TPB, 0, stream>>>(W2b, bn2b, hA, hB, nullptr, nullptr, 64, 64, MTK, 0);
  max_over_k<<<maxBlk, TPB, 0, stream>>>(hB, x123, 64);

  // ---- stage 3 (kNN on x2, conv3 128->64 -> x3)
  knn_kernel<<<BN, TPB, 0, stream>>>(x123 + 64 * NN, (long)192 * NN, 64, idx);
  conv_edge_gemm<<<edgeBlk, TPB, 0, stream>>>(W3, bn3, x123 + 64 * NN, (long)192 * NN, idx, hA, 64, 128, 64);
  max_over_k<<<maxBlk, TPB, 0, stream>>>(hA, x123, 128);

  // ---- global feature: conv4 (1024x192) fused with max over n
  init_xmax<<<(BB * 1024 + TPB - 1) / TPB, TPB, 0, stream>>>(xmaxU);
  conv_x123_gemm<<<(64 * (BN / 16)) / 8, TPB, 0, stream>>>(W4, 192, 0, bn4, x123, 1024, 0, xmaxU, nullptr, nullptr);
  lv_kernel<<<(BB * 64 + TPB - 1) / TPB, TPB, 0, stream>>>(Wl, bnl, l, lv);
  bias5_kernel<<<(BB * 256 + TPB - 1) / TPB, TPB, 0, stream>>>(W5, xmaxU, lv, bias5);

  // ---- head: conv5 (192-ch GEMM + hoisted bias), conv6, conv7, conv8
  conv_x123_gemm<<<(16 * (BN / 16)) / 8, TPB, 0, stream>>>(W5, 1280, 1088, bn5, x123, 256, 1, nullptr, bias5, y1);
  conv_f16_gemm<<<panBN, TPB, 0, stream>>>(W6, bn6, y1, y2, nullptr, nullptr, 256, 256, BN, 0);
  conv_f16_gemm<<<panBN, TPB, 0, stream>>>(W7, bn7, y2, y3, nullptr, nullptr, 128, 256, BN, 0);
  conv_f16_gemm<<<panBN, TPB, 0, stream>>>(W8, nullptr, y3, nullptr, out, b8, 50, 128, BN, 1);
  (void)in_sizes; (void)n_in; (void)out_size; (void)ws_size;
}
